// GraphUnet_26405458935812
// MI455X (gfx1250) — matrix operand order repaired and checked by reading, not compiled, not run on hardware
//
#include <hip/hip_runtime.h>
#include <hip/hip_bf16.h>

typedef __attribute__((ext_vector_type(16))) _Float16 v16h;
typedef __attribute__((ext_vector_type(8)))  _Float16 v8h;
typedef __attribute__((ext_vector_type(8)))  float    v8f;

// ---------------------------------------------------------------------------
// Generic WMMA GEMM: C[n x d] = Ah[n x Kp] * Bt[d x Kp]^T (+bias), f16 in /
// f32 out. Ah row-major f16, Bt row-major f16 (i.e. B transposed). Kp must be
// a multiple of 32 and zero-padded. One wave per 16x16 tile; inner loop is
// 4 x global_load_b128 + v_wmma, no branches.
// A-fragment layout (ISA 16-bit 16x32): lane m=lane&15, halves
//   [kb8..kb8+7] then [16+kb8..16+kb8+7], kb8 = (lane&16)?8:0  -> two b128s.
// B-fragment layout: lane col=lane&15, same k order as A
//   [kb8..kb8+7] then [16+kb8..16+kb8+7] -> two b128s.
// ---------------------------------------------------------------------------
__global__ void wmma_gemm_tn(const _Float16* __restrict__ Ah, const _Float16* __restrict__ Bt,
                             const float* __restrict__ bias, float* __restrict__ C,
                             int n, int d, int Kp) {
  const int lane = threadIdx.x;
  const int col0 = blockIdx.x * 16;
  const int row0 = blockIdx.y * 16;
  int arow = row0 + (lane & 15);
  if (arow > n - 1) arow = n - 1;           // clamp; masked at store
  const int bcol = col0 + (lane & 15);
  const _Float16* ap = Ah + (size_t)arow * Kp + ((lane & 16) ? 8 : 0);
  const _Float16* bp = Bt + (size_t)bcol * Kp + ((lane & 16) ? 8 : 0);
  v8f acc = {};
  for (int kk = 0; kk < Kp; kk += 32) {
    const v8h a0 = *(const v8h*)(ap);
    const v8h a1 = *(const v8h*)(ap + 16);
    const v16h a = __builtin_shufflevector(a0, a1, 0, 1, 2, 3, 4, 5, 6, 7,
                                           8, 9, 10, 11, 12, 13, 14, 15);
    const v8h b0 = *(const v8h*)(bp);
    const v8h b1 = *(const v8h*)(bp + 16);
    const v16h b = __builtin_shufflevector(b0, b1, 0, 1, 2, 3, 4, 5, 6, 7,
                                           8, 9, 10, 11, 12, 13, 14, 15);
    acc = __builtin_amdgcn_wmma_f32_16x16x32_f16(false, a, false, b, (short)0, acc,
                                                 false, false);
    ap += 32;
    bp += 32;
  }
  const float bb = bias ? bias[bcol] : 0.0f;
#pragma unroll
  for (int v = 0; v < 8; ++v) {
    const int r = row0 + v + ((lane & 16) ? 8 : 0);
    if (r < n) C[(size_t)r * d + bcol] = acc[v] + bb;
  }
}

// ---------------------------------------------------------------------------
// Prep kernels (all branchless: clamped loads + selects)
// ---------------------------------------------------------------------------
__global__ void cvt_f16_kernel(const float* __restrict__ src, _Float16* __restrict__ dst,
                               size_t count) {
  const size_t t = (size_t)blockIdx.x * blockDim.x + threadIdx.x;
  if (t < count) dst[t] = (_Float16)src[t];
}

// Wt[c][kk] = W[kk][c]  (f32 [k x d] -> f16 [d x k])
__global__ void transpose_cvt_kernel(const float* __restrict__ W, _Float16* __restrict__ Wt,
                                     int k, int d) {
  const size_t t = (size_t)blockIdx.x * blockDim.x + threadIdx.x;
  if (t >= (size_t)k * d) return;
  const int c = (int)(t / k), kk = (int)(t % k);
  Wt[t] = (_Float16)W[(size_t)kk * d + c];
}

// Mh[i][j] = (j<n && (i==j || adj(i,j)!=0)) ? 1 : 0, for i<n, j<npad.
__global__ void build_mask_kernel(const float* __restrict__ Af, const unsigned char* __restrict__ Au,
                                  _Float16* __restrict__ Mh, int n, int npad) {
  const size_t t = (size_t)blockIdx.x * blockDim.x + threadIdx.x;
  if (t >= (size_t)n * npad) return;
  const int i = (int)(t / npad), j = (int)(t % npad);
  const int jc = j < n ? j : n - 1;
  bool edge;
  if (Af) edge = (Af[(size_t)i * n + jc] != 0.0f);
  else    edge = (Au[(size_t)i * n + jc] != 0);
  const bool on = (j < n) && (edge || i == j);
  Mh[t] = (_Float16)(on ? 1.0f : 0.0f);
}

// Zt[c][j] (c<zr=272, j<npad):  c<d: e_j*H[j][c];  c==d: e_j;  else 0.  j>=n: 0
__global__ void build_zt_kernel(const float* __restrict__ H, const float* __restrict__ bvec,
                                const float* __restrict__ bmax, _Float16* __restrict__ Zt,
                                int n, int npad, int d, int zr) {
  const size_t t = (size_t)blockIdx.x * blockDim.x + threadIdx.x;
  if (t >= (size_t)zr * npad) return;
  const int c = (int)(t / npad), j = (int)(t % npad);
  const int jc = j < n ? j : n - 1;
  const int cc = c < d ? c : d - 1;
  const float h = H[(size_t)jc * d + cc];          // unconditional clamped load
  const float e = expf(bvec[jc] - *bmax);
  float v;
  if (c < d)       v = e * h;
  else if (c == d) v = e;
  else             v = 0.0f;
  if (j >= n) v = 0.0f;
  Zt[t] = (_Float16)v;
}

// ---------------------------------------------------------------------------
// bvec[i] = dot(H[i,:], phi[:])  — one wave32 per row.
// ---------------------------------------------------------------------------
__global__ void rowdot_kernel(const float* __restrict__ H, const float* __restrict__ phi,
                              float* __restrict__ bvec, int n, int d) {
  const int row = blockIdx.x * (blockDim.x >> 5) + (threadIdx.x >> 5);
  const int lane = threadIdx.x & 31;
  if (row >= n) return;
  float s = 0.0f;
  for (int c = lane; c < d; c += 32) s += H[(size_t)row * d + c] * phi[c];
  for (int off = 16; off > 0; off >>= 1) s += __shfl_down(s, off, 32);
  if (lane == 0) bvec[row] = s;
}

__global__ void reduce_max_kernel(const float* __restrict__ bvec, float* __restrict__ bmax,
                                  int n) {
  __shared__ float sm[1024];
  const int t = threadIdx.x;
  float m = -3.0e38f;
  for (int i = t; i < n; i += blockDim.x) m = fmaxf(m, bvec[i]);
  sm[t] = m;
  __syncthreads();
  for (int s = blockDim.x >> 1; s > 0; s >>= 1) {
    if (t < (int)s) sm[t] = fmaxf(sm[t], sm[t + s]);
    __syncthreads();
  }
  if (t == 0) *bmax = sm[0];
}

// out[i,c] = relu(Y[i,c] / Y[i,d]) (+ skip)
__global__ void finalize_gat(const float* __restrict__ Y, const float* __restrict__ skip,
                             float* __restrict__ out, int n, int d, int zc) {
  const size_t t = (size_t)blockIdx.x * blockDim.x + threadIdx.x;
  if (t >= (size_t)n * d) return;
  const int i = (int)(t / d), c = (int)(t % d);
  const float denom = Y[(size_t)i * zc + d];
  float v = fmaxf(Y[(size_t)i * zc + c] / denom, 0.0f);
  if (skip) v += skip[t];
  out[t] = v;
}

// scores[i] = sigmoid((H[i,:]·w + b)/100) — one wave32 per row
__global__ void pool_score_kernel(const float* __restrict__ H, const float* __restrict__ w,
                                  const float* __restrict__ b, float* __restrict__ sc,
                                  int n, int d) {
  const int row = blockIdx.x * (blockDim.x >> 5) + (threadIdx.x >> 5);
  const int lane = threadIdx.x & 31;
  if (row >= n) return;
  float s = 0.0f;
  for (int c = lane; c < d; c += 32) s += H[(size_t)row * d + c] * w[c];
  for (int off = 16; off > 0; off >>= 1) s += __shfl_down(s, off, 32);
  if (lane == 0) {
    const float z = (s + b[0]) * 0.01f;
    sc[row] = 1.0f / (1.0f + expf(-z));
  }
}

// Single-block bitonic top-k: sort (score desc, idx asc on ties), emit first keep.
__global__ void topk_sort_kernel(const float* __restrict__ sc, int n, int P, int keep,
                                 int* __restrict__ idx_out, float* __restrict__ val_out) {
  __shared__ float key[4096];
  __shared__ int   vid[4096];
  const int t = threadIdx.x;
  for (int i = t; i < P; i += blockDim.x) {
    key[i] = (i < n) ? sc[i] : -3.0e38f;
    vid[i] = i;
  }
  __syncthreads();
  for (int ksz = 2; ksz <= P; ksz <<= 1) {
    for (int j = ksz >> 1; j > 0; j >>= 1) {
      for (int i = t; i < P; i += blockDim.x) {
        const int l = i ^ j;
        if (l > i) {
          const bool desc = ((i & ksz) == 0);
          const float ki = key[i], kl = key[l];
          const int vi = vid[i], vl = vid[l];
          const bool l_before_i = (kl > ki) || (kl == ki && vl < vi);
          if (desc ? l_before_i : !l_before_i) {
            key[i] = kl; key[l] = ki; vid[i] = vl; vid[l] = vi;
          }
        }
      }
      __syncthreads();
    }
  }
  for (int i = t; i < keep; i += blockDim.x) { idx_out[i] = vid[i]; val_out[i] = key[i]; }
}

__global__ void gather_rows_kernel(const float* __restrict__ Xin, const int* __restrict__ idx,
                                   const float* __restrict__ vals, float* __restrict__ Xout,
                                   int nk, int d) {
  const size_t t = (size_t)blockIdx.x * blockDim.x + threadIdx.x;
  if (t >= (size_t)nk * d) return;
  const int i = (int)(t / d), c = (int)(t % d);
  Xout[t] = Xin[(size_t)idx[i] * d + c] * vals[i];
}

__global__ void gather_adj_kernel(const float* __restrict__ Af, const unsigned char* __restrict__ Au,
                                  const int* __restrict__ idx, unsigned char* __restrict__ Aout,
                                  int n_old, int nk) {
  const size_t t = (size_t)blockIdx.x * blockDim.x + threadIdx.x;
  if (t >= (size_t)nk * nk) return;
  const int i = (int)(t / nk), j = (int)(t % nk);
  const size_t src = (size_t)idx[i] * n_old + idx[j];
  Aout[t] = Af ? (unsigned char)(Af[src] != 0.0f) : (unsigned char)(Au[src] != 0);
}

__global__ void zero_f32_kernel(float* __restrict__ p, size_t count) {
  const size_t t = (size_t)blockIdx.x * blockDim.x + threadIdx.x;
  if (t < count) p[t] = 0.0f;
}

__global__ void scatter_rows_kernel(const float* __restrict__ Xs, const int* __restrict__ idx,
                                    float* __restrict__ Xbig, int nk, int d) {
  const size_t t = (size_t)blockIdx.x * blockDim.x + threadIdx.x;
  if (t >= (size_t)nk * d) return;
  const int i = (int)(t / d), c = (int)(t % d);
  Xbig[(size_t)idx[i] * d + c] = Xs[t];
}

__global__ void concat_kernel(const float* __restrict__ Xl, const float* __restrict__ Xr,
                              float* __restrict__ out, int n, int d) {
  const size_t t = (size_t)blockIdx.x * blockDim.x + threadIdx.x;
  if (t >= (size_t)n * 2 * d) return;
  const int i = (int)(t / (2 * d)), c = (int)(t % (2 * d));
  out[t] = (c < d) ? Xl[(size_t)i * d + c] : Xr[(size_t)i * d + (c - d)];
}

// ---------------------------------------------------------------------------
extern "C" void kernel_launch(void* const* d_in, const int* in_sizes, int n_in,
                              void* d_out, int out_size, void* d_ws, size_t ws_size,
                              hipStream_t stream) {
  (void)in_sizes; (void)n_in; (void)out_size; (void)ws_size;

  const float* A          = (const float*)d_in[0];
  const float* X          = (const float*)d_in[1];
  const float* start_W    = (const float*)d_in[2];
  const float* start_b    = (const float*)d_in[3];
  const float* start_phi  = (const float*)d_in[4];
  const float* down_W     = (const float*)d_in[5];
  const float* down_b     = (const float*)d_in[6];
  const float* down_phi   = (const float*)d_in[7];
  const float* pool_W     = (const float*)d_in[8];
  const float* pool_b     = (const float*)d_in[9];
  const float* bottom_W   = (const float*)d_in[10];
  const float* bottom_b   = (const float*)d_in[11];
  const float* bottom_phi = (const float*)d_in[12];
  const float* up_W       = (const float*)d_in[13];
  const float* up_b       = (const float*)d_in[14];
  const float* up_phi     = (const float*)d_in[15];
  const float* end_W      = (const float*)d_in[16];
  const float* end_b      = (const float*)d_in[17];
  const float* end_phi    = (const float*)d_in[18];

  const int D = 256, ZC = 272, N0 = 4096;
  const int ns[5] = {4096, 3686, 2580, 1548, 774};   // graph sizes per level
  const int P2[4] = {4096, 4096, 4096, 2048};        // bitonic pad sizes

  char* wp = (char*)d_ws;
  auto alloc = [&](size_t bytes) -> void* {
    void* r = (void*)wp;
    wp += (bytes + 255) & ~(size_t)255;
    return r;
  };

  float*         H      = (float*)alloc((size_t)N0 * D * 4);
  float*         bvec   = (float*)alloc((size_t)N0 * 4);
  float*         bmax   = (float*)alloc(256);
  float*         scores = (float*)alloc((size_t)N0 * 4);
  float*         vals   = (float*)alloc((size_t)N0 * 4);
  float*         Y      = (float*)alloc((size_t)N0 * ZC * 4);
  float*         Xa     = (float*)alloc((size_t)N0 * D * 4);
  float*         Xb     = (float*)alloc((size_t)N0 * D * 4);
  float*         Xcat   = (float*)alloc((size_t)N0 * 2 * D * 4);
  _Float16*      Xh     = (_Float16*)alloc((size_t)N0 * 2 * D * 2);  // f16 activations
  _Float16*      Wt     = (_Float16*)alloc((size_t)2 * D * D * 2);   // f16 W^T
  _Float16*      Zt     = (_Float16*)alloc((size_t)ZC * N0 * 2);     // f16 Z^T
  _Float16*      Mh     = (_Float16*)alloc((size_t)N0 * N0 * 2);     // f16 mask
  float*         downs[4];
  int*           idxs[4];
  unsigned char* Asub[4];
  for (int i = 0; i < 4; ++i) downs[i] = (float*)alloc((size_t)N0 * D * 4);
  for (int i = 0; i < 4; ++i) idxs[i]  = (int*)alloc((size_t)N0 * 4);
  for (int i = 0; i < 4; ++i) Asub[i]  = (unsigned char*)alloc((size_t)ns[i + 1] * ns[i + 1]);

  float* org = (float*)d_out + (size_t)N0 * D;  // second output of the tuple

  // One GAT layer. Softmax row-shift a_i cancels; global-shifted exp(b) gives
  // Y = M @ [e*H, e], out = relu(Y[:, :d]/Y[:, d]).
  auto gat = [&](const float* Af, const unsigned char* Au, int n,
                 const float* Xin, int kdim, const float* W, const float* bias,
                 const float* phi, float* out, const float* skip) {
    const int npad = (n + 31) & ~31;
    // ---- H = Xin @ W + bias (WMMA) ----
    const size_t cx = (size_t)n * kdim;
    cvt_f16_kernel<<<(unsigned)((cx + 255) / 256), 256, 0, stream>>>(Xin, Xh, cx);
    const size_t cw = (size_t)kdim * D;
    transpose_cvt_kernel<<<(unsigned)((cw + 255) / 256), 256, 0, stream>>>(W, Wt, kdim, D);
    dim3 g1(D / 16, (n + 15) / 16);
    wmma_gemm_tn<<<g1, 32, 0, stream>>>(Xh, Wt, bias, H, n, D, kdim);
    // ---- attention pieces ----
    rowdot_kernel<<<(n + 7) / 8, 256, 0, stream>>>(H, phi + D, bvec, n, D);
    reduce_max_kernel<<<1, 1024, 0, stream>>>(bvec, bmax, n);
    const size_t tz = (size_t)ZC * npad;
    build_zt_kernel<<<(unsigned)((tz + 255) / 256), 256, 0, stream>>>(H, bvec, bmax, Zt,
                                                                      n, npad, D, ZC);
    const size_t tm = (size_t)n * npad;
    build_mask_kernel<<<(unsigned)((tm + 255) / 256), 256, 0, stream>>>(Af, Au, Mh, n, npad);
    // ---- Y = M @ Z (WMMA) ----
    dim3 g2(ZC / 16, (n + 15) / 16);
    wmma_gemm_tn<<<g2, 32, 0, stream>>>(Mh, Zt, nullptr, Y, n, ZC, npad);
    const size_t to = (size_t)n * D;
    finalize_gat<<<(unsigned)((to + 255) / 256), 256, 0, stream>>>(Y, skip, out, n, D, ZC);
  };

  // ---- start GAT: X1 (= org) ----
  gat(A, nullptr, N0, X, D, start_W, start_b, start_phi, org, nullptr);

  // ---- down path with top-k pooling ----
  const float*         curAf = A;
  const unsigned char* curAu = nullptr;
  const float*         xin   = org;
  float* pp[2] = {Xa, Xb};
  for (int i = 0; i < 4; ++i) {
    const int n = ns[i], keep = ns[i + 1];
    gat(curAf, curAu, n, xin, D, down_W + (size_t)i * D * D, down_b + i * D,
        down_phi + i * 2 * D, downs[i], nullptr);
    pool_score_kernel<<<(n + 7) / 8, 256, 0, stream>>>(downs[i], pool_W + i * D,
                                                       pool_b + i, scores, n, D);
    topk_sort_kernel<<<1, 1024, 0, stream>>>(scores, n, P2[i], keep, idxs[i], vals);
    float* xp = pp[i & 1];
    const size_t tg = (size_t)keep * D;
    gather_rows_kernel<<<(unsigned)((tg + 255) / 256), 256, 0, stream>>>(downs[i], idxs[i],
                                                                         vals, xp, keep, D);
    const size_t ta = (size_t)keep * keep;
    gather_adj_kernel<<<(unsigned)((ta + 255) / 256), 256, 0, stream>>>(curAf, curAu, idxs[i],
                                                                        Asub[i], n, keep);
    xin = xp; curAf = nullptr; curAu = Asub[i];
  }

  // ---- bottom GAT (774 nodes), xin == Xb ----
  gat(nullptr, Asub[3], ns[4], xin, D, bottom_W, bottom_b, bottom_phi, Xa, nullptr);

  // ---- up path: unpool + GAT + skip ----
  float* Xc = Xa;
  float* other = Xb;
  for (int i = 0; i < 4; ++i) {
    const int j = 3 - i;
    const int nb = ns[j], cnt = ns[j + 1];
    const size_t tb = (size_t)nb * D;
    zero_f32_kernel<<<(unsigned)((tb + 255) / 256), 256, 0, stream>>>(other, tb);
    const size_t tsc = (size_t)cnt * D;
    scatter_rows_kernel<<<(unsigned)((tsc + 255) / 256), 256, 0, stream>>>(Xc, idxs[j], other,
                                                                           cnt, D);
    const float*         Af = (j == 0) ? A : nullptr;
    const unsigned char* Au = (j == 0) ? nullptr : Asub[j - 1];
    gat(Af, Au, nb, other, D, up_W + (size_t)i * D * D, up_b + i * D, up_phi + i * 2 * D,
        other, downs[j]);
    float* tmp = Xc; Xc = other; other = tmp;
  }

  // ---- end GAT on concat([X_c, org]) -> d_out[:N*D] ----
  const size_t tc = (size_t)N0 * 2 * D;
  concat_kernel<<<(unsigned)((tc + 255) / 256), 256, 0, stream>>>(Xc, org, Xcat, N0, D);
  gat(A, nullptr, N0, Xcat, 2 * D, end_W, end_b, end_phi, (float*)d_out, nullptr);
}
